// StandardMultiHeadAttention_10857677324500
// MI455X (gfx1250) — matrix operand order repaired and checked by reading, not compiled, not run on hardware
//
#include <hip/hip_runtime.h>
#include <math.h>

// ---------------------------------------------------------------------------
// Problem constants (match reference)
// ---------------------------------------------------------------------------
#define EMBED   1024
#define NHEADS  16
#define HDIM    64
#define BATCH   4
#define SEQ     2048
#define NROWS   (BATCH * SEQ)          // 8192
#define SCALE_F 0.125f                 // 1/sqrt(64)

// ---------------------------------------------------------------------------
// WMMA types (gfx1250, wave32)
// ---------------------------------------------------------------------------
typedef __attribute__((ext_vector_type(16))) __bf16 v16bf;
typedef __attribute__((ext_vector_type(8)))  float  v8f;
typedef int v4i __attribute__((__vector_size__(4 * sizeof(int))));

// Address-space-typed pointers for the async LDS-DMA builtin
typedef __attribute__((address_space(1))) v4i* gv4i_p;   // global
typedef __attribute__((address_space(3))) v4i* lv4i_p;   // LDS

union BF16Frag {
    v16bf          v;
    unsigned short s[16];
    uint4          q[2];
};

__device__ __forceinline__ unsigned short f2bf(float f) {
    unsigned int u = __float_as_uint(f);
    u += 0x7FFFu + ((u >> 16) & 1u);     // round-to-nearest-even
    return (unsigned short)(u >> 16);
}

__device__ __forceinline__ v8f wmma_bf16(const v16bf& a, const v16bf& b, const v8f& c) {
    return __builtin_amdgcn_wmma_f32_16x16x32_bf16(
        /*neg_a=*/false, a, /*neg_b=*/false, b,
        /*c_mod=*/(short)0, c, /*reuse_a=*/false, /*reuse_b=*/false);
}

// Async global->LDS copy path (ASYNCcnt), guarded: falls back to manual copy.
#if defined(__has_builtin)
#if __has_builtin(__builtin_amdgcn_global_load_async_to_lds_b128)
#define HAVE_ASYNC_LDS 1
#endif
#endif
#ifndef HAVE_ASYNC_LDS
#define HAVE_ASYNC_LDS 0
#endif

// ---------------------------------------------------------------------------
// Kernel 0: one-shot fp32 -> bf16 conversion (8 elements / thread, b128 I/O)
// ---------------------------------------------------------------------------
__global__ __launch_bounds__(256) void f32_to_bf16_kernel(
    const float* __restrict__ src, unsigned short* __restrict__ dst, int n8)
{
    const int i = blockIdx.x * blockDim.x + threadIdx.x;
    if (i >= n8) return;
    const float4* s = (const float4*)src;
    const float4 a = s[2 * i];
    const float4 b = s[2 * i + 1];
    uint4 o;
    o.x = (unsigned)f2bf(a.x) | ((unsigned)f2bf(a.y) << 16);
    o.y = (unsigned)f2bf(a.z) | ((unsigned)f2bf(a.w) << 16);
    o.z = (unsigned)f2bf(b.x) | ((unsigned)f2bf(b.y) << 16);
    o.w = (unsigned)f2bf(b.z) | ((unsigned)f2bf(b.w) << 16);
    ((uint4*)dst)[i] = o;
}

// ---------------------------------------------------------------------------
// Shared GEMM main loop: 256 threads (8 waves), block tile 128x128,
// wave tile 32x64 (2 m-subtiles x 4 n-subtiles), K-step 32.
// A, B are bf16 (u16 bits), both row-major with k contiguous (ld = EMBED).
// acc layout: acc[mt][nt], D-layout per CDNA5 ISA.
// ---------------------------------------------------------------------------
__device__ __forceinline__ void gemm_mainloop_bf16(
    const unsigned short* __restrict__ Ag,   // already offset to rowBase
    const unsigned short* __restrict__ Bg,   // already offset to fBase
    unsigned short* lds_a, unsigned short* lds_b,
    v8f (&acc)[2][4], int tid)
{
    const int lane = tid & 31;
    const int wave = tid >> 5;
    const int wm   = (wave >> 1) * 32;       // wave row offset (0..96)
    const int wn   = (wave & 1) * 64;        // wave col offset (0 or 64)
    const int hi   = lane >> 4;
    const int l16  = lane & 15;

    // LDS fill mapping: each thread moves 32B (16 u16) of one row
    const int frow = tid >> 1;               // 0..127
    const int fcol = (tid & 1) * 16;         // 0 or 16

    for (int k0 = 0; k0 < EMBED; k0 += 32) {
        const unsigned short* ap = Ag + (size_t)frow * EMBED + k0 + fcol;
        const unsigned short* bp = Bg + (size_t)frow * EMBED + k0 + fcol;
        unsigned short* la = &lds_a[frow * 32 + fcol];
        unsigned short* lb = &lds_b[frow * 32 + fcol];

#if HAVE_ASYNC_LDS
        __builtin_amdgcn_global_load_async_to_lds_b128((gv4i_p)ap,       (lv4i_p)la,       0, 0);
        __builtin_amdgcn_global_load_async_to_lds_b128((gv4i_p)(ap + 8), (lv4i_p)(la + 8), 0, 0);
        __builtin_amdgcn_global_load_async_to_lds_b128((gv4i_p)bp,       (lv4i_p)lb,       0, 0);
        __builtin_amdgcn_global_load_async_to_lds_b128((gv4i_p)(bp + 8), (lv4i_p)(lb + 8), 0, 0);
        asm volatile("s_wait_asynccnt 0x0" ::: "memory");
#else
        *(uint4*)la       = *(const uint4*)ap;
        *(uint4*)(la + 8) = *(const uint4*)(ap + 8);
        *(uint4*)lb       = *(const uint4*)bp;
        *(uint4*)(lb + 8) = *(const uint4*)(bp + 8);
#endif
        if (k0 + 32 < EMBED) {               // prefetch next K tile
            __builtin_prefetch(ap + 32, 0, 1);
            __builtin_prefetch(bp + 32, 0, 1);
        }
        __syncthreads();

        BF16Frag a[2], b[4];
#pragma unroll
        for (int t = 0; t < 2; ++t) {
            // A frag: row M = l16, K = (i<8?i:i+8) + 8*hi
            const unsigned short* arow = &lds_a[(wm + t * 16 + l16) * 32];
            a[t].q[0] = *(const uint4*)(arow + 8 * hi);
            a[t].q[1] = *(const uint4*)(arow + 16 + 8 * hi);
        }
#pragma unroll
        for (int t = 0; t < 4; ++t) {
            // B frag: col N = l16, K = (i<8?i:i+8) + 8*hi  (lds_b stored [n][k])
            const unsigned short* brow = &lds_b[(wn + t * 16 + l16) * 32];
            b[t].q[0] = *(const uint4*)(brow + 8 * hi);
            b[t].q[1] = *(const uint4*)(brow + 16 + 8 * hi);
        }
#pragma unroll
        for (int mt = 0; mt < 2; ++mt)
#pragma unroll
            for (int nt = 0; nt < 4; ++nt)
                acc[mt][nt] = wmma_bf16(a[mt].v, b[nt].v, acc[mt][nt]);
        __syncthreads();
    }
}

// ---------------------------------------------------------------------------
// Kernel 1: fused QKV projection (bf16 x, bf16 W -> bf16 Q/K/V in (B,H,N,Dh))
// Grid: (24 = 3 mats x 8 col tiles, 64 row tiles), 256 threads.
// ---------------------------------------------------------------------------
__global__ __launch_bounds__(256) void qkv_proj_kernel(
    const unsigned short* __restrict__ xb,
    const unsigned short* __restrict__ Wq,
    const unsigned short* __restrict__ Wk,
    const unsigned short* __restrict__ Wv,
    unsigned short* __restrict__ Q,
    unsigned short* __restrict__ K,
    unsigned short* __restrict__ V)
{
    __shared__ __attribute__((aligned(16))) unsigned short lds_a[128 * 32];
    __shared__ __attribute__((aligned(16))) unsigned short lds_b[128 * 32];

    const int tid     = threadIdx.x;
    const int bx      = blockIdx.x;             // 0..23
    const int which   = bx >> 3;                // 0=Q 1=K 2=V
    const int fBase   = (bx & 7) * 128;
    const int rowBase = blockIdx.y * 128;

    const unsigned short* W = (which == 0) ? Wq : ((which == 1) ? Wk : Wv);
    unsigned short* Out = (which == 0) ? Q : ((which == 1) ? K : V);

    v8f acc[2][4] = {};
    gemm_mainloop_bf16(xb + (size_t)rowBase * EMBED, W + (size_t)fBase * EMBED,
                       lds_a, lds_b, acc, tid);

    const int lane = tid & 31;
    const int wave = tid >> 5;
    const int wm = (wave >> 1) * 32, wn = (wave & 1) * 64;
    const int hi = lane >> 4, l16 = lane & 15;

#pragma unroll
    for (int mt = 0; mt < 2; ++mt)
#pragma unroll
        for (int nt = 0; nt < 4; ++nt)
#pragma unroll
            for (int r = 0; r < 8; ++r) {
                const int grow = rowBase + wm + mt * 16 + r + 8 * hi; // b*N+n
                const int gcol = fBase   + wn + nt * 16 + l16;        // h*64+d
                const int bb = grow >> 11;
                const int nn = grow & (SEQ - 1);
                const int hh = gcol >> 6;
                const int dd = gcol & 63;
                const size_t addr =
                    (((size_t)(bb * NHEADS + hh)) * SEQ + nn) * HDIM + dd;
                Out[addr] = f2bf(acc[mt][nt][r]);
            }
}

// ---------------------------------------------------------------------------
// Kernel 2: causal flash attention, one wave per 16-query tile.
// ---------------------------------------------------------------------------
__global__ __launch_bounds__(32) void flash_attn_kernel(
    const unsigned short* __restrict__ Q,
    const unsigned short* __restrict__ K,
    const unsigned short* __restrict__ V,
    unsigned short* __restrict__ O)
{
    __shared__ __attribute__((aligned(16))) unsigned short pbuf[16 * 32];

    const int lane  = threadIdx.x & 31;
    const int hi    = lane >> 4;
    const int l16   = lane & 15;
    const int qt    = blockIdx.x;
    const int h     = blockIdx.y;
    const int b     = blockIdx.z;
    const int qBase = qt * 16;

    const size_t headOff = ((size_t)(b * NHEADS + h)) * SEQ * HDIM;
    const unsigned short* Qb = Q + headOff;
    const unsigned short* Kb = K + headOff;
    const unsigned short* Vb = V + headOff;

    BF16Frag qf[2];
#pragma unroll
    for (int t = 0; t < 2; ++t) {
        const unsigned short* row = Qb + (size_t)(qBase + l16) * HDIM + t * 32;
        qf[t].q[0] = *(const uint4*)(row + 8 * hi);
        qf[t].q[1] = *(const uint4*)(row + 16 + 8 * hi);
    }

    float mrow[8], lrow[8];
    v8f acc[4] = {};
#pragma unroll
    for (int r = 0; r < 8; ++r) { mrow[r] = -INFINITY; lrow[r] = 0.f; }

    const int nkb = (qBase + 16 + 31) >> 5;
    for (int kb = 0; kb < nkb; ++kb) {
        const int keyBase = kb * 32;

        v8f sh[2];
#pragma unroll
        for (int nh = 0; nh < 2; ++nh) {
            v8f s = {};
#pragma unroll
            for (int t = 0; t < 2; ++t) {
                BF16Frag kf;   // B frag: N = key, K = d = t*32 + (i<8?i:i+8) + 8*hi
                const unsigned short* row =
                    Kb + (size_t)(keyBase + nh * 16 + l16) * HDIM + t * 32 + 8 * hi;
                kf.q[0] = *(const uint4*)(row);
                kf.q[1] = *(const uint4*)(row + 16);
                s = wmma_bf16(qf[t].v, kf.v, s);
            }
            sh[nh] = s;
        }

        const bool needMask = (keyBase + 31) > qBase;
#pragma unroll
        for (int nh = 0; nh < 2; ++nh)
#pragma unroll
            for (int r = 0; r < 8; ++r) {
                float v = sh[nh][r] * SCALE_F;
                if (needMask) {
                    const int key  = keyBase + nh * 16 + l16;
                    const int qrow = qBase + r + 8 * hi;
                    if (key > qrow) v = -1e30f;
                }
                sh[nh][r] = v;
            }

        float corr[8];
#pragma unroll
        for (int r = 0; r < 8; ++r) {
            float t = fmaxf(sh[0][r], sh[1][r]);
#pragma unroll
            for (int m = 8; m >= 1; m >>= 1) t = fmaxf(t, __shfl_xor(t, m, 32));
            const float mnew = fmaxf(mrow[r], t);
            corr[r] = __expf(mrow[r] - mnew);
            mrow[r] = mnew;
            const float p0 = __expf(sh[0][r] - mnew);
            const float p1 = __expf(sh[1][r] - mnew);
            sh[0][r] = p0; sh[1][r] = p1;
            float rs = p0 + p1;
#pragma unroll
            for (int m = 8; m >= 1; m >>= 1) rs += __shfl_xor(rs, m, 32);
            lrow[r] = lrow[r] * corr[r] + rs;
        }
#pragma unroll
        for (int t = 0; t < 4; ++t)
#pragma unroll
            for (int r = 0; r < 8; ++r) acc[t][r] *= corr[r];

        __syncthreads();
#pragma unroll
        for (int nh = 0; nh < 2; ++nh)
#pragma unroll
            for (int r = 0; r < 8; ++r)
                pbuf[(r + 8 * hi) * 32 + nh * 16 + l16] = f2bf(sh[nh][r]);
        __syncthreads();

        BF16Frag pf;
        {
            const unsigned short* prow = &pbuf[l16 * 32];
            pf.q[0] = *(const uint4*)(prow + 8 * hi);
            pf.q[1] = *(const uint4*)(prow + 16 + 8 * hi);
        }

#pragma unroll
        for (int t = 0; t < 4; ++t) {
            BF16Frag vf;   // B frag: N = d, K = key = (i<8?i:i+8) + 8*hi
#pragma unroll
            for (int i = 0; i < 16; ++i) {
                const int key = keyBase + (i < 8 ? i : i + 8) + 8 * hi;
                vf.s[i] = Vb[(size_t)key * HDIM + t * 16 + l16];
            }
            acc[t] = wmma_bf16(pf.v, vf.v, acc[t]);
        }
    }

#pragma unroll
    for (int t = 0; t < 4; ++t)
#pragma unroll
        for (int r = 0; r < 8; ++r) {
            const int qrow = qBase + r + 8 * hi;
            const float val = acc[t][r] / lrow[r];
            const size_t addr =
                ((size_t)(b * SEQ + qrow)) * EMBED + h * HDIM + t * 16 + l16;
            O[addr] = f2bf(val);
        }
}

// ---------------------------------------------------------------------------
// Kernel 3: output projection (bf16 O, bf16 Wo -> fp32 out)
// Grid: (8 col tiles, 64 row tiles), 256 threads.
// ---------------------------------------------------------------------------
__global__ __launch_bounds__(256) void out_proj_kernel(
    const unsigned short* __restrict__ Oin,
    const unsigned short* __restrict__ Wo,
    float* __restrict__ out)
{
    __shared__ __attribute__((aligned(16))) unsigned short lds_a[128 * 32];
    __shared__ __attribute__((aligned(16))) unsigned short lds_b[128 * 32];

    const int tid     = threadIdx.x;
    const int fBase   = blockIdx.x * 128;
    const int rowBase = blockIdx.y * 128;

    v8f acc[2][4] = {};
    gemm_mainloop_bf16(Oin + (size_t)rowBase * EMBED, Wo + (size_t)fBase * EMBED,
                       lds_a, lds_b, acc, tid);

    const int lane = tid & 31;
    const int wave = tid >> 5;
    const int wm = (wave >> 1) * 32, wn = (wave & 1) * 64;
    const int hi = lane >> 4, l16 = lane & 15;

#pragma unroll
    for (int mt = 0; mt < 2; ++mt)
#pragma unroll
        for (int nt = 0; nt < 4; ++nt)
#pragma unroll
            for (int r = 0; r < 8; ++r) {
                const int grow = rowBase + wm + mt * 16 + r + 8 * hi;
                const int gcol = fBase   + wn + nt * 16 + l16;
                out[(size_t)grow * EMBED + gcol] = acc[mt][nt][r];
            }
}

// ---------------------------------------------------------------------------
// Host launcher
// ---------------------------------------------------------------------------
extern "C" void kernel_launch(void* const* d_in, const int* in_sizes, int n_in,
                              void* d_out, int out_size, void* d_ws, size_t ws_size,
                              hipStream_t stream) {
    const float* x  = (const float*)d_in[0];
    // d_in[1] = mask (int32 tril) — causal mask hardcoded in flash kernel
    const float* Wq = (const float*)d_in[2];
    const float* Wk = (const float*)d_in[3];
    const float* Wv = (const float*)d_in[4];
    const float* Wo = (const float*)d_in[5];
    float* out = (float*)d_out;

    // Workspace layout (u16 bf16):
    //   Q | K | V | O (each NROWS*EMBED) | xb (NROWS*EMBED) | wq|wk|wv|wo (each EMBED^2)
    const size_t elems = (size_t)NROWS * EMBED;   // 8M
    const size_t welems = (size_t)EMBED * EMBED;  // 1M
    unsigned short* qws = (unsigned short*)d_ws;
    unsigned short* kws = qws + elems;
    unsigned short* vws = kws + elems;
    unsigned short* ows = vws + elems;
    unsigned short* xb  = ows + elems;
    unsigned short* wqb = xb + elems;
    unsigned short* wkb = wqb + welems;
    unsigned short* wvb = wkb + welems;
    unsigned short* wob = wvb + welems;

    // 0) one-shot bf16 conversions
    {
        const int n8x = (int)(elems / 8);
        f32_to_bf16_kernel<<<(n8x + 255) / 256, 256, 0, stream>>>(x, xb, n8x);
        const int n8w = (int)(welems / 8);
        f32_to_bf16_kernel<<<(n8w + 255) / 256, 256, 0, stream>>>(Wq, wqb, n8w);
        f32_to_bf16_kernel<<<(n8w + 255) / 256, 256, 0, stream>>>(Wk, wkb, n8w);
        f32_to_bf16_kernel<<<(n8w + 255) / 256, 256, 0, stream>>>(Wv, wvb, n8w);
        f32_to_bf16_kernel<<<(n8w + 255) / 256, 256, 0, stream>>>(Wo, wob, n8w);
    }

    // 1) QKV projection: 24 col tiles (3 mats x 8) x 64 row tiles
    qkv_proj_kernel<<<dim3(24, 64), 256, 0, stream>>>(xb, wqb, wkb, wvb, qws, kws, vws);

    // 2) Causal flash attention
    flash_attn_kernel<<<dim3(SEQ / 16, NHEADS, BATCH), 32, 0, stream>>>(qws, kws, vws, ows);

    // 3) Output projection
    out_proj_kernel<<<dim3(8, 64), 256, 0, stream>>>(ows, wob, out);
}
